// PILOT_IPAdapterAttnProcessor_12163347382378
// MI455X (gfx1250) — compile-verified, matrix operand order checked by reading, not run on hardware
//
#include <hip/hip_runtime.h>
#include <hip/hip_bf16.h>

typedef __attribute__((ext_vector_type(16))) __bf16 v16bf;
typedef __attribute__((ext_vector_type(8)))  float  v8f;
typedef __attribute__((ext_vector_type(4)))  int    vi4;

union Acc { v8f v; float f[8]; };

#if defined(__HIP_DEVICE_COMPILE__) && \
    __has_builtin(__builtin_amdgcn_global_load_async_to_lds_b128) && \
    __has_builtin(__builtin_amdgcn_s_wait_asynccnt)
#define USE_ASYNC_COPY 1
#else
#define USE_ASYNC_COPY 0
#endif

__device__ inline void async_copy16(const void* g, void* l) {
#if USE_ASYNC_COPY
  __builtin_amdgcn_global_load_async_to_lds_b128(
      (__attribute__((address_space(1))) vi4*)g,
      (__attribute__((address_space(3))) vi4*)l, 0, 0);
#endif
}
__device__ inline void async_wait() {
#if USE_ASYNC_COPY
  __builtin_amdgcn_s_wait_asynccnt(0);
#endif
}

__device__ inline v8f wmma_bf16(v16bf a, v16bf b, v8f c) {
  return __builtin_amdgcn_wmma_f32_16x16x32_bf16(false, a, false, b, (short)0, c, false, false);
}

// Build a 16x32 bf16 fragment (A layout: lane holds row = row_base + lane%16;
// K pairs per ISA: kk = (j<4?0:16) + 8*(lane/16) + 2*(j&3)).  B fragments use
// the same builder because we store B transposed in LDS as [n][k].
__device__ inline v16bf make_frag(const __bf16* lds, int stride, int row_base, int k_base) {
  int lane = threadIdx.x & 31;
  int m  = lane & 15;
  int h2 = lane >> 4;
  union { v16bf v; unsigned int u[8]; } f;
  const __bf16* rowp = lds + (size_t)(row_base + m) * stride + k_base;
#pragma unroll
  for (int j = 0; j < 8; ++j) {
    int kk = ((j < 4) ? 0 : 16) + h2 * 8 + (j & 3) * 2;
    f.u[j] = *(const unsigned int*)(rowp + kk);
  }
  return f.v;
}

__device__ inline float red16_max(float v) {
#pragma unroll
  for (int m = 1; m < 16; m <<= 1) v = fmaxf(v, __shfl_xor(v, m, 32));
  return v;
}
__device__ inline float red16_sum(float v) {
#pragma unroll
  for (int m = 1; m < 16; m <<= 1) v += __shfl_xor(v, m, 32);
  return v;
}

// ---------------------------------------------------------------- cast kernel
__global__ void cast_f32_bf16(const float* __restrict__ in, __bf16* __restrict__ out, int n) {
  for (int i = blockIdx.x * blockDim.x + threadIdx.x; i < n; i += gridDim.x * blockDim.x)
    out[i] = (__bf16)in[i];
}

// ------------------------------------------------------------ tiled WMMA GEMM
// C[M,N] = A[M,K] * B[K,N]; A,B bf16 row-major.  K % 32 == 0, N % 128 == 0.
// mode 0: Cb (bf16) = result.   mode 1: Cf (f32) = result + bias[col] + extra[row,col].
__global__ __launch_bounds__(256) void gemm_bf16(
    const __bf16* __restrict__ A, const __bf16* __restrict__ Bm,
    int M, int N, int K,
    __bf16* __restrict__ Cb, float* __restrict__ Cf,
    const float* __restrict__ bias, const float* __restrict__ extra, int mode) {
  __shared__ __align__(16) __bf16 a_sh[128 * 40];   // [m][k], stride 40 (80B, 16B-aligned)
  __shared__ __align__(16) __bf16 b_sh[128 * 40];   // [n][k], stride 40 (B transposed in LDS)

  const int tid  = threadIdx.x;
  const int col0 = blockIdx.x * 128;
  const int row0 = blockIdx.y * 128;
  const int wave = tid >> 5, lane = tid & 31;
  const int wm = wave & 3;    // 4 waves along M (32 rows each)
  const int wn = wave >> 2;   // 2 waves along N (64 cols each)
  constexpr bool kAsync = (USE_ASYNC_COPY != 0);
  const bool interior = (row0 + 128 <= M);

  Acc acc[2][4];
#pragma unroll
  for (int a = 0; a < 2; ++a)
#pragma unroll
    for (int b = 0; b < 4; ++b)
#pragma unroll
      for (int r = 0; r < 8; ++r) acc[a][b].f[r] = 0.f;

  const uint4 zero4 = make_uint4(0u, 0u, 0u, 0u);

  for (int k0 = 0; k0 < K; k0 += 32) {
    // ---- stage A tile: 128x32 bf16 as 512 16B chunks (2 per thread)
    if (kAsync && interior) {
      for (int i = tid; i < 512; i += 256) {
        int r = i >> 2, c4 = i & 3;
        async_copy16(A + (size_t)(row0 + r) * K + k0 + c4 * 8, &a_sh[r * 40 + c4 * 8]);
      }
    } else {
      for (int i = tid; i < 512; i += 256) {
        int r = i >> 2, c4 = i & 3;
        int gr = row0 + r;
        uint4 v = zero4;
        if (gr < M) v = *(const uint4*)(A + (size_t)gr * K + k0 + c4 * 8);
        *(uint4*)&a_sh[r * 40 + c4 * 8] = v;
      }
    }
    // ---- stage B tile transposed: thread owns (k-pair, 8-col chunk);
    //      two row loads, 8 interleaved b32 stores (contiguous K pairs in LDS).
    {
      int kp = tid >> 4;    // 0..15 -> k = 2kp, 2kp+1
      int nc = tid & 15;    // 0..15 -> cols nc*8 .. nc*8+7
      const __bf16* bp = Bm + (size_t)(k0 + 2 * kp) * N + col0 + nc * 8;
      uint4 r0 = *(const uint4*)bp;
      uint4 r1 = *(const uint4*)(bp + N);
      const unsigned short* p0 = (const unsigned short*)&r0;
      const unsigned short* p1 = (const unsigned short*)&r1;
#pragma unroll
      for (int j = 0; j < 8; ++j) {
        unsigned int pk = (unsigned int)p0[j] | ((unsigned int)p1[j] << 16);
        *(unsigned int*)&b_sh[(nc * 8 + j) * 40 + 2 * kp] = pk;
      }
    }
    async_wait();
    __syncthreads();

    if (k0 + 32 < K) {  // speculative prefetch of next B tile -> global_prefetch_b8
      __builtin_prefetch(&Bm[(size_t)(k0 + 32 + (tid >> 3)) * N + col0 + ((tid & 7) << 4)], 0, 0);
    }

    v16bf afr[2];
    afr[0] = make_frag(a_sh, 40, wm * 32 + 0, 0);
    afr[1] = make_frag(a_sh, 40, wm * 32 + 16, 0);
#pragma unroll
    for (int nt = 0; nt < 4; ++nt) {
      v16bf bfr = make_frag(b_sh, 40, wn * 64 + nt * 16, 0);
      acc[0][nt].v = wmma_bf16(afr[0], bfr, acc[0][nt].v);
      acc[1][nt].v = wmma_bf16(afr[1], bfr, acc[1][nt].v);
    }
    __syncthreads();
  }

#pragma unroll
  for (int mt = 0; mt < 2; ++mt)
#pragma unroll
    for (int nt = 0; nt < 4; ++nt)
#pragma unroll
      for (int r = 0; r < 8; ++r) {
        int row = row0 + wm * 32 + mt * 16 + (lane >> 4) * 8 + r;
        int col = col0 + wn * 64 + nt * 16 + (lane & 15);
        if (row < M) {
          if (mode == 0) {
            Cb[(size_t)row * N + col] = (__bf16)acc[mt][nt].f[r];
          } else {
            float v = acc[mt][nt].f[r] + bias[col] + extra[(size_t)row * N + col];
            Cf[(size_t)row * N + col] = v;
          }
        }
      }
}

// ------------------------------------------------------------ fused attention
// grid (Sq/64, H, B), 128 threads (4 waves).  Wave w owns q rows [w*16, w*16+16).
// Writes attn_out (bf16, goes through Wo) and ip_out (f32, added after Wo).
__global__ __launch_bounds__(128) void attn_wmma(
    const __bf16* __restrict__ Q, const __bf16* __restrict__ Kt,
    const __bf16* __restrict__ Vt, const __bf16* __restrict__ Kip,
    const __bf16* __restrict__ Vip, const float* __restrict__ mask,
    __bf16* __restrict__ attn_out, float* __restrict__ ip_out) {
  const int Cdim = 1280, Sq = 4096, T = 77, NIP = 16;
  const int q0 = blockIdx.x * 64;
  const int hd = blockIdx.y;
  const int b  = blockIdx.z;
  constexpr bool kAsync = (USE_ASYNC_COPY != 0);

  __shared__ __align__(16) __bf16 q_sh[64 * 72];     // [q][d]
  __shared__ __align__(16) __bf16 k_sh[96 * 72];     // [t][d], rows >=77 zero
  __shared__ __align__(16) __bf16 vt_sh[64 * 100];   // [d][t], t >=77 zero
  __shared__ __align__(16) __bf16 p_sh[64 * 100];    // [q][t] probs (bf16)
  __shared__ __align__(16) __bf16 kip_sh[16 * 72];   // [t][d]
  __shared__ __align__(16) __bf16 vipt_sh[64 * 36];  // [d][t], t>=16 zero
  __shared__ __align__(16) __bf16 pip_sh[64 * 36];   // [q][t], t>=16 zero

  const int tid = threadIdx.x;
  const uint4 zero4 = make_uint4(0u, 0u, 0u, 0u);

  // Q: 64x64, pure vector copy (scale applied in softmax instead)
  for (int i = tid; i < 512; i += 128) {
    int r = i >> 3, c8 = i & 7;
    const __bf16* gp = Q + (size_t)(b * Sq + q0 + r) * Cdim + hd * 64 + c8 * 8;
    __bf16* lp = &q_sh[r * 72 + c8 * 8];
    if (kAsync) async_copy16(gp, lp);
    else        *(uint4*)lp = *(const uint4*)gp;
  }
  // K: 96x64, rows >= 77 zero
  for (int i = tid; i < 768; i += 128) {
    int t = i >> 3, c8 = i & 7;
    __bf16* lp = &k_sh[t * 72 + c8 * 8];
    if (t < T) {
      const __bf16* gp = Kt + (size_t)(b * T + t) * Cdim + hd * 64 + c8 * 8;
      if (kAsync) async_copy16(gp, lp);
      else        *(uint4*)lp = *(const uint4*)gp;
    } else {
      *(uint4*)lp = zero4;
    }
  }
  // Kip: 16x64
  {
    int t = tid >> 3, c8 = tid & 7;
    const __bf16* gp = Kip + (size_t)(b * NIP + t) * Cdim + hd * 64 + c8 * 8;
    __bf16* lp = &kip_sh[t * 72 + c8 * 8];
    if (kAsync) async_copy16(gp, lp);
    else        *(uint4*)lp = *(const uint4*)gp;
  }
  // V transposed (coalesced global reads, scattered b16 LDS stores)
  for (int i = tid; i < 96 * 64; i += 128) {
    int t = i >> 6, c = i & 63;
    vt_sh[c * 100 + t] = (t < T) ? Vt[(size_t)(b * T + t) * Cdim + hd * 64 + c] : (__bf16)0.0f;
  }
  for (int i = tid; i < 64 * 32; i += 128) {
    int c = i >> 5, t = i & 31;
    vipt_sh[c * 36 + t] = (t < NIP) ? Vip[(size_t)(b * NIP + t) * Cdim + hd * 64 + c] : (__bf16)0.0f;
  }
  async_wait();
  __syncthreads();

  const int wave = tid >> 5, lane = tid & 31;
  const int nco = lane & 15, hh = lane >> 4;
  const int qb = wave * 16;

  // ---- scores S = Q @ K^T : 6 n-tiles x 2 k-steps
  Acc s[6];
#pragma unroll
  for (int nt = 0; nt < 6; ++nt)
#pragma unroll
    for (int r = 0; r < 8; ++r) s[nt].f[r] = 0.f;
#pragma unroll
  for (int ks = 0; ks < 2; ++ks) {
    v16bf aq = make_frag(q_sh, 72, qb, ks * 32);
#pragma unroll
    for (int nt = 0; nt < 6; ++nt) {
      v16bf bk = make_frag(k_sh, 72, nt * 16, ks * 32);
      s[nt].v = wmma_bf16(aq, bk, s[nt].v);
    }
  }

  // ---- row softmax (scale folded here) + post-softmax mask, P -> LDS bf16
  const float NEGI = -1e30f;
  const float scale = 0.125f;   // 1/sqrt(64)
#pragma unroll
  for (int r = 0; r < 8; ++r) {
    float mx = NEGI;
#pragma unroll
    for (int nt = 0; nt < 6; ++nt) {
      int j = nt * 16 + nco;
      float v = s[nt].f[r] * scale;
      s[nt].f[r] = v;
      mx = fmaxf(mx, (j < T) ? v : NEGI);
    }
    mx = red16_max(mx);
    float sum = 0.f;
#pragma unroll
    for (int nt = 0; nt < 6; ++nt) {
      int j = nt * 16 + nco;
      float e = (j < T) ? __expf(s[nt].f[r] - mx) : 0.f;
      s[nt].f[r] = e;
      sum += e;
    }
    sum = red16_sum(sum);
    float inv = 1.f / sum;
    int qrow = qb + hh * 8 + r;
    int gq = q0 + qrow;
#pragma unroll
    for (int nt = 0; nt < 6; ++nt) {
      int j = nt * 16 + nco;
      float p = s[nt].f[r] * inv;
      p = (j < T) ? p * mask[(size_t)gq * T + j] : 0.f;
      p_sh[qrow * 100 + j] = (__bf16)p;
    }
  }
  __syncthreads();

  // ---- O = P @ V : 4 n-tiles (d) x 3 k-steps (96 padded keys)
  Acc o[4];
#pragma unroll
  for (int nt = 0; nt < 4; ++nt)
#pragma unroll
    for (int r = 0; r < 8; ++r) o[nt].f[r] = 0.f;
#pragma unroll
  for (int ks = 0; ks < 3; ++ks) {
    v16bf ap = make_frag(p_sh, 100, qb, ks * 32);
#pragma unroll
    for (int nt = 0; nt < 4; ++nt) {
      v16bf bv = make_frag(vt_sh, 100, nt * 16, ks * 32);
      o[nt].v = wmma_bf16(ap, bv, o[nt].v);
    }
  }

  // ---- IP stream: 16 tokens, one score tile
  Acc sip;
#pragma unroll
  for (int r = 0; r < 8; ++r) sip.f[r] = 0.f;
#pragma unroll
  for (int ks = 0; ks < 2; ++ks) {
    v16bf aq = make_frag(q_sh, 72, qb, ks * 32);
    v16bf bk = make_frag(kip_sh, 72, 0, ks * 32);
    sip.v = wmma_bf16(aq, bk, sip.v);
  }
#pragma unroll
  for (int r = 0; r < 8; ++r) {
    float v = sip.f[r] * scale;
    float mx = red16_max(v);
    float e = __expf(v - mx);
    float sum = red16_sum(e);
    int qrow = qb + hh * 8 + r;
    pip_sh[qrow * 36 + nco] = (__bf16)(e / sum);
    pip_sh[qrow * 36 + 16 + nco] = (__bf16)0.0f;   // zero pad cols 16..31
  }
  __syncthreads();

  Acc oip[4];
#pragma unroll
  for (int nt = 0; nt < 4; ++nt)
#pragma unroll
    for (int r = 0; r < 8; ++r) oip[nt].f[r] = 0.f;
  {
    v16bf ap = make_frag(pip_sh, 36, qb, 0);
#pragma unroll
    for (int nt = 0; nt < 4; ++nt) {
      v16bf bv = make_frag(vipt_sh, 36, nt * 16, 0);
      oip[nt].v = wmma_bf16(ap, bv, oip[nt].v);
    }
  }

  // ---- write outputs
#pragma unroll
  for (int nt = 0; nt < 4; ++nt)
#pragma unroll
    for (int r = 0; r < 8; ++r) {
      int qrow = qb + hh * 8 + r;
      size_t row = (size_t)(b * Sq + q0 + qrow);
      int col = hd * 64 + nt * 16 + nco;
      attn_out[row * Cdim + col] = (__bf16)o[nt].f[r];
      ip_out[row * Cdim + col] = oip[nt].f[r];
    }
}

// ------------------------------------------------------------------- launcher
extern "C" void kernel_launch(void* const* d_in, const int* in_sizes, int n_in,
                              void* d_out, int out_size, void* d_ws, size_t ws_size,
                              hipStream_t stream) {
  (void)in_sizes; (void)n_in; (void)out_size; (void)ws_size;
  const float* hs   = (const float*)d_in[0];
  const float* enc  = (const float*)d_in[1];
  const float* iph  = (const float*)d_in[2];
  const float* mask = (const float*)d_in[3];
  const float* Wq   = (const float*)d_in[4];
  const float* Wk   = (const float*)d_in[5];
  const float* Wv   = (const float*)d_in[6];
  const float* Wo   = (const float*)d_in[7];
  const float* bo   = (const float*)d_in[8];
  const float* Wkip = (const float*)d_in[9];
  const float* Wvip = (const float*)d_in[10];
  float* out = (float*)d_out;

  const int Bn = 4, Sq = 4096, T = 77, Cdim = 1280, Dd = 2048, NIP = 16, H = 20;
  const int Mq = Bn * Sq;     // 16384
  const int Mkv = Bn * T;     // 308
  const int Mip = Bn * NIP;   // 64

  char* p = (char*)d_ws;
  auto carve = [&](size_t bytes) -> void* {
    void* r = (void*)p;
    p += (bytes + 255) & ~(size_t)255;
    return r;
  };
  __bf16* hsb   = (__bf16*)carve((size_t)Mq * Cdim * 2);
  __bf16* encb  = (__bf16*)carve((size_t)Mkv * Dd * 2);
  __bf16* ipb   = (__bf16*)carve((size_t)Mip * Dd * 2);
  __bf16* wqb   = (__bf16*)carve((size_t)Cdim * Cdim * 2);
  __bf16* wkb   = (__bf16*)carve((size_t)Dd * Cdim * 2);
  __bf16* wvb   = (__bf16*)carve((size_t)Dd * Cdim * 2);
  __bf16* wob   = (__bf16*)carve((size_t)Cdim * Cdim * 2);
  __bf16* wkipb = (__bf16*)carve((size_t)Dd * Cdim * 2);
  __bf16* wvipb = (__bf16*)carve((size_t)Dd * Cdim * 2);
  __bf16* Qb    = (__bf16*)carve((size_t)Mq * Cdim * 2);
  __bf16* Kb    = (__bf16*)carve((size_t)Mkv * Cdim * 2);
  __bf16* Vb    = (__bf16*)carve((size_t)Mkv * Cdim * 2);
  __bf16* Kipb  = (__bf16*)carve((size_t)Mip * Cdim * 2);
  __bf16* Vipb  = (__bf16*)carve((size_t)Mip * Cdim * 2);
  __bf16* AOb   = (__bf16*)carve((size_t)Mq * Cdim * 2);
  float*  IPO   = (float*)carve((size_t)Mq * Cdim * 4);

  auto cast = [&](const float* src, __bf16* dst, int n) {
    int blocks = (n + 255) / 256;
    if (blocks > 2048) blocks = 2048;
    cast_f32_bf16<<<blocks, 256, 0, stream>>>(src, dst, n);
  };
  cast(hs, hsb, Mq * Cdim);
  cast(enc, encb, Mkv * Dd);
  cast(iph, ipb, Mip * Dd);
  cast(Wq, wqb, Cdim * Cdim);
  cast(Wk, wkb, Dd * Cdim);
  cast(Wv, wvb, Dd * Cdim);
  cast(Wo, wob, Cdim * Cdim);
  cast(Wkip, wkipb, Dd * Cdim);
  cast(Wvip, wvipb, Dd * Cdim);

  dim3 gQ(Cdim / 128, (Mq + 127) / 128);
  gemm_bf16<<<gQ, 256, 0, stream>>>(hsb, wqb, Mq, Cdim, Cdim, Qb, nullptr, nullptr, nullptr, 0);

  dim3 gKV(Cdim / 128, (Mkv + 127) / 128);
  gemm_bf16<<<gKV, 256, 0, stream>>>(encb, wkb, Mkv, Cdim, Dd, Kb, nullptr, nullptr, nullptr, 0);
  gemm_bf16<<<gKV, 256, 0, stream>>>(encb, wvb, Mkv, Cdim, Dd, Vb, nullptr, nullptr, nullptr, 0);

  dim3 gIP(Cdim / 128, (Mip + 127) / 128);
  gemm_bf16<<<gIP, 256, 0, stream>>>(ipb, wkipb, Mip, Cdim, Dd, Kipb, nullptr, nullptr, nullptr, 0);
  gemm_bf16<<<gIP, 256, 0, stream>>>(ipb, wvipb, Mip, Cdim, Dd, Vipb, nullptr, nullptr, nullptr, 0);

  dim3 gA(Sq / 64, H, Bn);
  attn_wmma<<<gA, 128, 0, stream>>>(Qb, Kb, Vb, Kipb, Vipb, mask, AOb, IPO);

  dim3 gO(Cdim / 128, Mq / 128);
  gemm_bf16<<<gO, 256, 0, stream>>>(AOb, wob, Mq, Cdim, Cdim, nullptr, out, bo, IPO, 1);
}
